// NSSNN_43078521979476
// MI455X (gfx1250) — compile-verified, matrix operand order checked by reading, not run on hardware
//
#include <hip/hip_runtime.h>
#include <hip/hip_bf16.h>

// ---------------------------------------------------------------------------
// NSSNN forward for MI455X (gfx1250): conv3d layers as implicit GEMM on
// v_wmma_f32_16x16x32_bf16 (wave32), bf16 input slabs double-buffered in LDS
// via GLOBAL_LOAD_ASYNC_TO_LDS_B32 (ASYNCcnt pipeline).
// ---------------------------------------------------------------------------

#define D_ 31
#define H_ 96
#define W_ 96
#define V_ (D_*H_*W_)
#define HW_ (H_*W_)

typedef __attribute__((ext_vector_type(16))) __bf16           v16bf;
typedef __attribute__((ext_vector_type(16))) unsigned short   v16us;
typedef __attribute__((ext_vector_type(8)))  float            v8f;

#if __has_builtin(__builtin_amdgcn_global_load_async_to_lds_b32) && \
    __has_builtin(__builtin_amdgcn_s_wait_asynccnt)
#define HAVE_ASYNC_LDS 1
#else
#define HAVE_ASYNC_LDS 0
#endif

template <bool B> struct BoolTag { static constexpr bool value = B; };

__device__ __forceinline__ unsigned short f2bf(float f) {
    unsigned u = __float_as_uint(f);
    unsigned r = (u + 0x7FFFu + ((u >> 16) & 1u)) >> 16;   // RNE bf16
    return (unsigned short)r;
}

#if HAVE_ASYNC_LDS
typedef __attribute__((address_space(1))) int gint_t;   // global
typedef __attribute__((address_space(3))) int lint_t;   // LDS

__device__ __forceinline__ void async_cp_b32(const unsigned short* gsrc,
                                             unsigned short* lds_dst) {
    __builtin_amdgcn_global_load_async_to_lds_b32(
        (gint_t*)gsrc, (lint_t*)lds_dst, 0, 0);
}
#endif

// ---------------------------------------------------------------------------
// float -> bf16 layer-input conversion (done once per conv layer)
// ---------------------------------------------------------------------------
__global__ void tobf_kernel(const float* __restrict__ in,
                            unsigned short* __restrict__ ob, long long n)
{
    long long i = blockIdx.x * (long long)blockDim.x + threadIdx.x;
    if (i < n) ob[i] = f2bf(in[i]);
}

// ---------------------------------------------------------------------------
// K ordering: k = blk*KB + kw*Cin + ci, blk = kd*3+kh (9 blocks), KB = Cin*3
// padded to a multiple of 32; Cin is always a power of two here, so
// ci = l & (Cin-1), kw = l >> log2(Cin).
// A_pack[((tile*nChunk + c)*32 + lane)*16 + e] holds the 16x32 bf16 WMMA A
// fragment in per-lane element order
//   kk(e,lane) = (e&7) + ((e>>3)<<4) + ((lane>>4)<<3),  M = lane&15.
// deconv=1 applies flip + (O,I) transpose: w_eff[co][ci][k27] =
// w_orig[ci][co][26-k27].
// ---------------------------------------------------------------------------
__global__ void pack_w_kernel(const float* __restrict__ w,
                              unsigned short* __restrict__ pack,
                              int Cout, int Cin, int cpb, int deconv, int total,
                              int kwShift, int ciMask)
{
    int idx = blockIdx.x * blockDim.x + threadIdx.x;
    if (idx >= total) return;
    int e      = idx & 15;
    int lane   = (idx >> 4) & 31;
    int rest   = idx >> 9;             // tile*nChunk + c
    int nChunk = 9 * cpb;
    int c      = rest % nChunk;
    int tile   = rest / nChunk;
    int kk     = (e & 7) + ((e >> 3) << 4) + ((lane >> 4) << 3);
    int blkI   = c / cpb;
    int cb     = c - blkI * cpb;
    int l      = cb * 32 + kk;         // local K within (kd,kh) block
    int kd     = blkI / 3;
    int kh     = blkI - kd * 3;
    int ci     = l & ciMask;
    int kw     = l >> kwShift;
    int co     = tile * 16 + (lane & 15);
    float v = 0.f;
    if (kw < 3 && co < Cout) {
        int ko = (kd * 3 + kh) * 3 + kw;
        if (deconv) v = w[(ci * Cout + co) * 27 + (26 - ko)];
        else        v = w[(co * Cin + ci) * 27 + ko];
    }
    pack[idx] = f2bf(v);
}

// ---------------------------------------------------------------------------
// Implicit-GEMM conv3d, pad=1, stride=1, 3x3x3.
// Workgroup = 4 waves at one (d,h): wave = (coGroup<<1)|colHalf.
// Each wave: 4 co-tiles x 3 col-subtiles (48 cols) -> 12 WMMA per K-chunk
// off 3 shared B fragments.  Input slab (bf16, W-halo padded, rows of 100)
//   slab[ci][col] = xb[ci, d+kd-1, h+kh-1, col-2]
// is double-buffered in LDS and streamed with async global->LDS while the
// previous block is consumed.  Out-of-volume (kd,kh) rows are skipped.
// Hot layers take the branch-free FULL path (all 4 co-tiles active).
// ---------------------------------------------------------------------------
__global__ __launch_bounds__(128)
void conv_wmma_kernel(const unsigned short* __restrict__ xb,
                      const unsigned short* __restrict__ packA,
                      float* __restrict__ out,
                      int Cin, int Cout, int cpb, int kwShift, int ciMask)
{
    const int tid     = threadIdx.x;
    const int lane    = tid & 31;
    const int wave    = tid >> 5;
    const int colHalf = wave & 1;          // cols [colHalf*48, +48)
    const int coGrp   = wave >> 1;         // 4 co-tiles per group
    const int d       = blockIdx.y;
    const int h       = blockIdx.z;
    const int hi      = lane >> 4;
    const int lm      = lane & 15;
    const int nChunk  = 9 * cpb;
    const int tiles   = (Cout + 15) >> 4;
    const int tbase   = (blockIdx.x * 2 + coGrp) * 4;
    const int nt      = tiles - tbase;     // active co-tiles (t < 4 bound below)

    __shared__ __align__(16) unsigned short slab[2][64 * 100];   // 2 x 12.8 KB

    v8f acc[4][3] = {};

    // W-halo columns (col 1 = x[-1], col 98 = x[96]) are always zero; async
    // staging never touches them, so write once for both buffers.
    for (int idx = tid; idx < 2 * Cin; idx += 128) {
        int ci   = idx >> 1;
        int colz = (idx & 1) ? 98 : 1;
        slab[0][ci * 100 + colz] = 0;
        slab[1][ci * 100 + colz] = 0;
    }

    // valid (kd,kh) block list for this (d,h)
    int vb[9]; int nv = 0;
    #pragma unroll
    for (int b = 0; b < 9; ++b) {
        int kd = b / 3, kh = b - kd * 3;
        int zz = d + kd - 1, yy = h + kh - 1;
        if (zz >= 0 && zz < D_ && yy >= 0 && yy < H_) vb[nv++] = b;
    }

    __syncthreads();                       // halo zeros visible

    auto stage = [&](int b, int buf) {     // issue async copies for block b
        int kd = b / 3, kh = b - kd * 3;
        int zz = d + kd - 1, yy = h + kh - 1;
        const unsigned short* src = xb + (zz * H_ + yy) * W_;
        unsigned short* sl = &slab[buf][0];
        const int nCp = Cin * 48;          // b32 copies, 2 bf16 each
        for (int idx = tid; idx < nCp; idx += 128) {
            int ci = idx / 48;
            int j  = idx - ci * 48;
#if HAVE_ASYNC_LDS
            async_cp_b32(src + ci * V_ + 2 * j, sl + ci * 100 + 2 + 2 * j);
#else
            *reinterpret_cast<unsigned int*>(sl + ci * 100 + 2 + 2 * j) =
                *reinterpret_cast<const unsigned int*>(src + ci * V_ + 2 * j);
#endif
        }
    };

    if (nv > 0) stage(vb[0], 0);

    for (int i = 0; i < nv; ++i) {
#if HAVE_ASYNC_LDS
        __builtin_amdgcn_s_wait_asynccnt(0);   // slab i fully landed
#endif
        __syncthreads();                        // all waves' staging visible
        if (i + 1 < nv) stage(vb[i + 1], (i + 1) & 1);  // overlap with compute

        const unsigned short* sl = &slab[i & 1][0];
        const int blkI = vb[i];

        auto chunk_loop = [&](auto fullTag) {
            constexpr bool FULL = decltype(fullTag)::value;
            // strength-reduced per-tile A pointers (advance 512 per chunk)
            const unsigned short* ap[4];
            #pragma unroll
            for (int t = 0; t < 4; ++t)
                ap[t] = packA + (size_t)((tbase + t) * nChunk + blkI * cpb) * 512
                              + lane * 16;
            for (int cb = 0; cb < cpb; ++cb) {
                v16bf a[4];
                #pragma unroll
                for (int t = 0; t < 4; ++t) {
                    if (FULL || t < nt) {
                        a[t] = __builtin_bit_cast(v16bf,
                                  *reinterpret_cast<const v16us*>(ap[t]));
                        __builtin_prefetch(ap[t] + 512, 0, 0);
                        ap[t] += 512;
                    }
                }
                // per-element slab offsets: depend on cb/hi only, shared by
                // all 3 subtiles
                const int lbase = cb * 32 + (hi << 3);
                int offs[16];
                #pragma unroll
                for (int e = 0; e < 16; ++e) {
                    int l  = lbase + (e & 7) + ((e >> 3) << 4);
                    int ci = l & ciMask;
                    int kw = l >> kwShift;
                    kw = kw > 2 ? 2 : kw;      // padded K: packed A is zero
                    offs[e] = ci * 100 + kw + 1;
                }
                #pragma unroll
                for (int ss = 0; ss < 3; ++ss) {
                    const int xcol = (colHalf * 3 + ss) * 16 + lm;
                    v16us bu;
                    #pragma unroll
                    for (int e = 0; e < 16; ++e)
                        bu[e] = sl[offs[e] + xcol];
                    v16bf b = __builtin_bit_cast(v16bf, bu);
                    #pragma unroll
                    for (int t = 0; t < 4; ++t)
                        if (FULL || t < nt)
                            acc[t][ss] = __builtin_amdgcn_wmma_f32_16x16x32_bf16(
                                false, a[t], false, b, (short)0, acc[t][ss], false, false);
                }
            }
        };

        if (nt >= 4)     chunk_loop(BoolTag<true>{});    // hot layers
        else if (nt > 0) chunk_loop(BoolTag<false>{});   // ragged tail
    }

    // D fragment: VGPR v -> M = v + 8*hi, N = lane&15
    const int rowbase = (d * H_ + h) * W_;
    if (nt >= 4 && (Cout & 15) == 0) {
        #pragma unroll
        for (int t = 0; t < 4; ++t) {
            #pragma unroll
            for (int ss = 0; ss < 3; ++ss) {
                int col = rowbase + (colHalf * 3 + ss) * 16 + lm;
                #pragma unroll
                for (int v = 0; v < 8; ++v) {
                    int co = (tbase + t) * 16 + v + 8 * hi;
                    out[co * V_ + col] = acc[t][ss][v];
                }
            }
        }
    } else {
        #pragma unroll
        for (int t = 0; t < 4; ++t) {
            if (t < nt) {
                #pragma unroll
                for (int ss = 0; ss < 3; ++ss) {
                    int col = rowbase + (colHalf * 3 + ss) * 16 + lm;
                    #pragma unroll
                    for (int v = 0; v < 8; ++v) {
                        int co = (tbase + t) * 16 + v + 8 * hi;
                        if (co < Cout) out[co * V_ + col] = acc[t][ss][v];
                    }
                }
            }
        }
    }
}

// ---------------------------------------------------------------------------
// Gate activations in-place on conv output g (nGroups groups of Ch channels):
// tanh on first and last group (Wx, X), sigmoid on the middle gates.
// ---------------------------------------------------------------------------
__global__ void act_kernel(float* __restrict__ g, int nGroups, int Ch, long long n)
{
    long long i = blockIdx.x * (long long)blockDim.x + threadIdx.x;
    if (i >= n) return;
    int group = (int)(i / ((long long)Ch * V_));
    float v = g[i];
    if (group == 0 || group == nGroups - 1) v = tanhf(v);
    else                                    v = 1.f / (1.f + __expf(-v));
    g[i] = v;
}

// ---------------------------------------------------------------------------
// SRU scan (one direction). Thread per (c,h,w) line; state C carried in a
// register over D.  i==0 uses wv=1 so C0 = 1-f0 (matches reference).
// accum=1 adds into out (reverse pass of bidirectional SRUs).
// ---------------------------------------------------------------------------
__global__ void sru_kernel(const float* __restrict__ X, const float* __restrict__ Wx,
                           const float* __restrict__ F, const float* __restrict__ R,
                           float* __restrict__ out, int Ch, int reverse, int accum)
{
    int t = blockIdx.x * blockDim.x + threadIdx.x;
    if (t >= Ch * HW_) return;
    int cidx = t / HW_;
    int s    = t - cidx * HW_;
    int base = cidx * V_ + s;
    float C = 0.f;
    for (int i = 0; i < D_; ++i) {
        int dd  = reverse ? (D_ - 1 - i) : i;
        int off = base + dd * HW_;
        float f  = F[off];
        float r  = R[off];
        float wv = (i == 0) ? 1.f : Wx[off];
        C = f * C + (1.f - f) * wv;
        float hv = r * C + (1.f - r) * X[off];
        if (accum) out[off] += hv; else out[off] = hv;
    }
}

__global__ void add_kernel(float* __restrict__ dst, const float* __restrict__ a,
                           const float* __restrict__ b, long long n)
{
    long long i = blockIdx.x * (long long)blockDim.x + threadIdx.x;
    if (i < n) dst[i] = a[i] + b[i];
}

// ---------------------------------------------------------------------------
static inline unsigned blk(long long n) { return (unsigned)((n + 255) / 256); }

static void run_conv(const float* xin, const float* w, float* g,
                     unsigned short* packbuf, unsigned short* xbf,
                     int Cin, int Cout, int deconv, hipStream_t stream)
{
    int cpb     = (Cin * 3 + 31) / 32;     // K-chunks per (kd,kh) block
    int nChunk  = 9 * cpb;
    int tiles   = (Cout + 15) / 16;
    int kwShift = __builtin_ctz((unsigned)Cin);
    int ciMask  = Cin - 1;
    long long n = (long long)Cin * V_;
    tobf_kernel<<<blk(n), 256, 0, stream>>>(xin, xbf, n);
    int total = tiles * nChunk * 512;
    pack_w_kernel<<<(total + 255) / 256, 256, 0, stream>>>(
        w, packbuf, Cout, Cin, cpb, deconv, total, kwShift, ciMask);
    dim3 grid((tiles + 7) / 8, D_, H_);
    conv_wmma_kernel<<<grid, 128, 0, stream>>>(
        xbf, packbuf, g, Cin, Cout, cpb, kwShift, ciMask);
}

extern "C" void kernel_launch(void* const* d_in, const int* in_sizes, int n_in,
                              void* d_out, int out_size, void* d_ws, size_t ws_size,
                              hipStream_t stream)
{
    (void)in_sizes; (void)n_in; (void)out_size; (void)ws_size;
    const float* x      = (const float*)d_in[0];
    const float* w_head = (const float*)d_in[1];
    const float* w_e0   = (const float*)d_in[2];
    const float* w_e1   = (const float*)d_in[3];
    const float* w_d0   = (const float*)d_in[4];
    const float* w_d1   = (const float*)d_in[5];
    const float* w_tail = (const float*)d_in[6];
    float* out = (float*)d_out;

    const long long V = V_;
    float* ws  = (float*)d_ws;
    float* g   = ws;                  // up to 256 channels of conv output
    float* h   = ws + 256 * V;        // 16 ch
    float* e0  = ws + 272 * V;        // 32 ch
    float* e1  = ws + 304 * V;        // 64 ch
    float* d0b = ws + 368 * V;        // 32 ch
    float* d1b = ws + 400 * V;        // 16 ch
    float* tmp = ws + 416 * V;        // 1 ch (tail SRU result)
    unsigned short* packbuf = (unsigned short*)(ws + 417 * V);
    unsigned short* xbf     = (unsigned short*)(ws + 418 * V);  // bf16 inputs

    // ---- head: h = do_sru(x, w_head)  (6 groups of 16) --------------------
    run_conv(x, w_head, g, packbuf, xbf, 1, 96, 0, stream);
    act_kernel<<<blk(96 * V), 256, 0, stream>>>(g, 6, 16, 96 * V);
    // split: Wx(0) ft(1) ft2(2) rt(3) rt2(4) X(5)
    sru_kernel<<<blk(16 * HW_), 256, 0, stream>>>(g + 5*16*V, g, g + 1*16*V, g + 3*16*V, h, 16, 0, 0);
    sru_kernel<<<blk(16 * HW_), 256, 0, stream>>>(g + 5*16*V, g, g + 2*16*V, g + 4*16*V, h, 16, 1, 1);

    // ---- e0 = si_sru(h, w_e0, fwd)  (4 groups of 32) ----------------------
    run_conv(h, w_e0, g, packbuf, xbf, 16, 128, 0, stream);
    act_kernel<<<blk(128 * V), 256, 0, stream>>>(g, 4, 32, 128 * V);
    sru_kernel<<<blk(32 * HW_), 256, 0, stream>>>(g + 3*32*V, g, g + 1*32*V, g + 2*32*V, e0, 32, 0, 0);

    // ---- e1 = si_sru(e0, w_e1, rev)  (4 groups of 64) ---------------------
    run_conv(e0, w_e1, g, packbuf, xbf, 32, 256, 0, stream);
    act_kernel<<<blk(256 * V), 256, 0, stream>>>(g, 4, 64, 256 * V);
    sru_kernel<<<blk(64 * HW_), 256, 0, stream>>>(g + 3*64*V, g, g + 1*64*V, g + 2*64*V, e1, 64, 1, 0);

    // ---- d0 = si_sru(e1, w_d0, fwd, deconv) + e0 --------------------------
    run_conv(e1, w_d0, g, packbuf, xbf, 64, 128, 1, stream);
    act_kernel<<<blk(128 * V), 256, 0, stream>>>(g, 4, 32, 128 * V);
    sru_kernel<<<blk(32 * HW_), 256, 0, stream>>>(g + 3*32*V, g, g + 1*32*V, g + 2*32*V, d0b, 32, 0, 0);
    add_kernel<<<blk(32 * V), 256, 0, stream>>>(d0b, d0b, e0, 32 * V);

    // ---- d1 = si_sru(d0, w_d1, rev, deconv) + h ---------------------------
    run_conv(d0b, w_d1, g, packbuf, xbf, 32, 64, 1, stream);
    act_kernel<<<blk(64 * V), 256, 0, stream>>>(g, 4, 16, 64 * V);
    sru_kernel<<<blk(16 * HW_), 256, 0, stream>>>(g + 3*16*V, g, g + 1*16*V, g + 2*16*V, d1b, 16, 1, 0);
    add_kernel<<<blk(16 * V), 256, 0, stream>>>(d1b, d1b, h, 16 * V);

    // ---- out = do_sru(d1, w_tail, deconv) + x  (6 groups of 1) ------------
    run_conv(d1b, w_tail, g, packbuf, xbf, 16, 6, 1, stream);
    act_kernel<<<blk(6 * V), 256, 0, stream>>>(g, 6, 1, 6 * V);
    sru_kernel<<<blk(HW_), 256, 0, stream>>>(g + 5*V, g, g + 1*V, g + 3*V, tmp, 1, 0, 0);
    sru_kernel<<<blk(HW_), 256, 0, stream>>>(g + 5*V, g, g + 2*V, g + 4*V, tmp, 1, 1, 1);
    add_kernel<<<blk(V), 256, 0, stream>>>(out, tmp, x, V);
}
